// Attention_52587579572920
// MI455X (gfx1250) — matrix operand order repaired and checked by reading, not compiled, not run on hardware
//
#include <hip/hip_runtime.h>
#include <stdint.h>

// ---------------------------------------------------------------------------
// CDNA5 (gfx1250) attention: bf16 WMMA pipeline with async-LDS staged GEMMs
//   cast -> QKV GEMM (async LDS + wmma) -> RoPE/pack -> flash attn (wmma)
//   -> O GEMM
// ---------------------------------------------------------------------------

typedef __bf16 bf16;
typedef __attribute__((ext_vector_type(16))) __bf16 v16bf;
typedef __attribute__((ext_vector_type(8)))  __bf16 v8bf;
typedef __attribute__((ext_vector_type(8)))  float  v8f;

#define S_LEN  2048
#define HDIM   4096
#define NHEADS 32
#define NKVH   8
#define HEADD  128
#define KVD    (NKVH * HEADD)   // 1024

// A-operand fragment (16x32 bf16): lane holds row m = lane&15, half h = lane>>4.
// elements 0..15 <-> K = h*16 + 0..15
static __device__ __forceinline__ v16bf load_a_frag(const bf16* row, int h) {
  v8bf lo = *(const v8bf*)(row + h * 16);
  v8bf hi = *(const v8bf*)(row + h * 16 + 8);
  v16bf a;
#pragma unroll
  for (int i = 0; i < 8; ++i) { a[i] = lo[i]; a[i + 8] = hi[i]; }
  return a;
}

static __device__ __forceinline__ v8f wmma_bf16(v16bf a, v16bf b, v8f c) {
  // (neg_a, A, neg_b, B, c_mod, C, reuse_a, reuse_b)
  return __builtin_amdgcn_wmma_f32_16x16x32_bf16(false, a, false, b,
                                                 (short)0, c, false, false);
}

// ------------------------------- cast --------------------------------------
__global__ void cast_f32_bf16(const float* __restrict__ in,
                              bf16* __restrict__ out, long n) {
  long i = (long)blockIdx.x * blockDim.x + threadIdx.x;
  if (i < n) out[i] = (bf16)in[i];
}

// ------------------------------- GEMM --------------------------------------
// C[M,N] (f32) = A[M,K] (bf16 row-major) * W[N,K]^T (bf16 row-major)
// Workgroup tile: 128(M) x 64(N); 8 waves, each 16(M) x 64(N).
// The shared 64x32 B-slab per K-step is staged through LDS with
// global_load_async_to_lds_b128 (double buffered, ASYNCcnt-tracked).
__global__ void __launch_bounds__(256)
gemm_bf16(const bf16* __restrict__ A, const bf16* __restrict__ W,
          float* __restrict__ C, int M, int N, int K) {
  __shared__ bf16 smB[2][64 * 32];   // 2 x 4KB double buffer
  const int tilesN = N >> 6;
  const int bm = blockIdx.x / tilesN;            // 128-row M tile
  const int bn = blockIdx.x % tilesN;            // 64-col  N tile
  const int wave = threadIdx.x >> 5;
  const int lane = threadIdx.x & 31;
  const int n16 = lane & 15, h = lane >> 4;

  // Async-copy assignment: each thread moves one 16B chunk of the B slab.
  const int brow  = threadIdx.x >> 2;            // 0..63  (N row of W)
  const int bchnk = threadIdx.x & 3;             // 0..3   (16B chunk in 64B row)
  const bf16* wsrc = W + (size_t)(bn * 64 + brow) * K + bchnk * 8;

  const bf16* arow = A + (size_t)(bm * 128 + wave * 16 + n16) * K;

  auto issue = [&](int buf, int kb) {
    unsigned lds = (unsigned)(uintptr_t)&smB[buf][brow * 32 + bchnk * 8];
    unsigned long long g = (unsigned long long)(uintptr_t)(wsrc + kb);
    asm volatile("global_load_async_to_lds_b128 %0, %1, off"
                 :: "v"(lds), "v"(g) : "memory");
  };

  v8f acc[4] = {};
  const int nsteps = K >> 5;
  issue(0, 0);
  for (int s = 0; s < nsteps; ++s) {
    const int kb = s << 5;
    if (s + 1 < nsteps) {
      issue((s + 1) & 1, kb + 32);               // prefetch next slab
      asm volatile("s_wait_asynccnt 1" ::: "memory");
    } else {
      asm volatile("s_wait_asynccnt 0" ::: "memory");
    }
    __syncthreads();                             // slab visible to all waves
    v16bf af = load_a_frag(arow + kb, h);
    const bf16* bb = &smB[s & 1][0];
#pragma unroll
    for (int c = 0; c < 4; ++c) {
      // B frag: 32B contiguous per lane from LDS (ds_load_b128 x2)
      v16bf bfr = *(const v16bf*)(bb + (c * 16 + n16) * 32 + h * 16);
      acc[c] = wmma_bf16(af, bfr, acc[c]);
    }
    __syncthreads();                             // reads done before reuse
  }
#pragma unroll
  for (int c = 0; c < 4; ++c)
#pragma unroll
    for (int r = 0; r < 8; ++r)
      C[(size_t)(bm * 128 + wave * 16 + r + 8 * h) * N +
        bn * 64 + c * 16 + n16] = acc[c][r];
}

// ------------------------- RoPE + head packing -----------------------------
// in : f32 [S][nh*HD]   out : bf16 [nh][S][HD]  (RoPE applied)
__global__ void rope_pack(const float* __restrict__ in,
                          bf16* __restrict__ out, int nh) {
  long idx = (long)blockIdx.x * blockDim.x + threadIdx.x;
  long total = (long)S_LEN * nh * HEADD;
  if (idx >= total) return;
  int d = (int)(idx & 127);
  long t = idx >> 7;
  int head = (int)(t % nh);
  int s = (int)(t / nh);
  int i2 = d & 63;
  // inv_freq = theta^(-2*i/128) = exp(-(2i/128)*ln(500000))
  float invf = __expf(-(float)(2 * i2) * (1.0f / 128.0f) * 13.122363377404328f);
  float ang = (float)s * invf;
  float sn, cs;
  __sincosf(ang, &sn, &cs);
  const float* base = in + (size_t)s * (nh * HEADD) + head * HEADD;
  float x0 = base[d];
  float rot = (d < 64) ? -base[d + 64] : base[d - 64];
  out[((size_t)head * S_LEN + s) * HEADD + d] = (bf16)(x0 * cs + rot * sn);
}

// V transpose: f32 [S][NKVH*HD] -> bf16 [NKVH][HD][S]
__global__ void v_transpose(const float* __restrict__ in,
                            bf16* __restrict__ out) {
  long idx = (long)blockIdx.x * blockDim.x + threadIdx.x;
  long total = (long)NKVH * HEADD * S_LEN;
  if (idx >= total) return;
  int s = (int)(idx % S_LEN);
  long t = idx / S_LEN;
  int d = (int)(t % HEADD);
  int kv = (int)(t / HEADD);
  out[idx] = (bf16)in[(size_t)s * KVD + kv * HEADD + d];
}

// --------------------------- flash attention -------------------------------
// One wave per (head, 16-query tile). 32 keys per iteration.
// Qr: bf16 [NH][S][HD], Kr: bf16 [NKVH][S][HD], VT: bf16 [NKVH][HD][S]
// O : bf16 [S][H]
__global__ void __launch_bounds__(256)
fa_kernel(const bf16* __restrict__ Qr, const bf16* __restrict__ Kr,
          const bf16* __restrict__ VT, bf16* __restrict__ O) {
  __shared__ bf16 sm[8][16 * 32];   // per-wave P staging (C-layout -> A-layout)
  const int wave = threadIdx.x >> 5;
  const int lane = threadIdx.x & 31;
  const int gid = blockIdx.x * 8 + wave;
  const int qt = gid & 127;          // S/16 = 128 query tiles
  const int head = gid >> 7;
  if (head >= NHEADS) return;
  const int kvh = head >> 2;         // GQA: 4 query heads per kv head
  const int n16 = lane & 15, h = lane >> 4;
  const float scale = 0.08838834764831845f;  // 1/sqrt(128)

  // Q tile 16x128 -> 4 A-fragments (K=32 each)
  const bf16* qbase = Qr + ((size_t)head * S_LEN + qt * 16 + n16) * HEADD;
  v16bf qa[4];
#pragma unroll
  for (int kk = 0; kk < 4; ++kk) qa[kk] = load_a_frag(qbase + kk * 32, h);

  const bf16* kbase = Kr + (size_t)kvh * S_LEN * HEADD;
  const bf16* vbase = VT + (size_t)kvh * HEADD * S_LEN;

  v8f o[8] = {};
  float m_run[8], l_run[8];
#pragma unroll
  for (int r = 0; r < 8; ++r) { m_run[r] = -1e30f; l_run[r] = 0.0f; }

  const int nb = (qt * 16 + 15) / 32 + 1;   // causal: key blocks needed
  for (int b = 0; b < nb; ++b) {
    const int kb = b * 32;
    // S = Q K^T  (two 16x16 score tiles for 32 keys)
    v8f s0 = {}, s1 = {};
#pragma unroll
    for (int kk = 0; kk < 4; ++kk) {
      v16bf b0 = *(const v16bf*)(kbase + (size_t)(kb + n16) * HEADD + kk * 32 + h * 16);
      s0 = wmma_bf16(qa[kk], b0, s0);
      v16bf b1 = *(const v16bf*)(kbase + (size_t)(kb + 16 + n16) * HEADD + kk * 32 + h * 16);
      s1 = wmma_bf16(qa[kk], b1, s1);
    }
    // scale + causal mask; C layout: row m = r + 8*h, col n = lane&15
    const int qrow = qt * 16 + 8 * h;
#pragma unroll
    for (int r = 0; r < 8; ++r) {
      float v0 = s0[r] * scale, v1 = s1[r] * scale;
      if (kb + n16 > qrow + r)      v0 = -1e30f;
      if (kb + 16 + n16 > qrow + r) v1 = -1e30f;
      s0[r] = v0; s1[r] = v1;
    }
    // online softmax; row reductions stay inside 16-lane halves (masks 1..8)
#pragma unroll
    for (int r = 0; r < 8; ++r) {
      float rm = fmaxf(s0[r], s1[r]);
#pragma unroll
      for (int msk = 1; msk < 16; msk <<= 1)
        rm = fmaxf(rm, __shfl_xor(rm, msk, 32));
      float mn = fmaxf(m_run[r], rm);
      float corr = __expf(m_run[r] - mn);
      m_run[r] = mn;
      float p0 = __expf(s0[r] - mn);
      float p1 = __expf(s1[r] - mn);
      float rs = p0 + p1;
#pragma unroll
      for (int msk = 1; msk < 16; msk <<= 1)
        rs += __shfl_xor(rs, msk, 32);
      l_run[r] = l_run[r] * corr + rs;
#pragma unroll
      for (int c = 0; c < 8; ++c) o[c][r] *= corr;
      // stage P into LDS at [row m][key] for the A-layout reload
      sm[wave][(r + 8 * h) * 32 + n16]      = (bf16)p0;
      sm[wave][(r + 8 * h) * 32 + 16 + n16] = (bf16)p1;
    }
    // reload P as A-operand (wave-private LDS region; DS ops in-order per wave)
    v16bf pA = load_a_frag(&sm[wave][n16 * 32], h);
    // O += P V  : B-operand rows = keys, contiguous along S in VT
#pragma unroll
    for (int c = 0; c < 8; ++c) {
      v16bf vb = *(const v16bf*)(vbase + (size_t)(c * 16 + n16) * S_LEN + kb + h * 16);
      o[c] = wmma_bf16(pA, vb, o[c]);
    }
  }
  // epilogue: normalize by row sum, write bf16 [S][H]
  bf16* obase = O + (size_t)(qt * 16 + 8 * h) * HDIM + head * HEADD + n16;
#pragma unroll
  for (int r = 0; r < 8; ++r) {
    float inv = 1.0f / l_run[r];
#pragma unroll
    for (int c = 0; c < 8; ++c)
      obase[(size_t)r * HDIM + c * 16] = (bf16)(o[c][r] * inv);
  }
}

// ------------------------------- launch ------------------------------------
extern "C" void kernel_launch(void* const* d_in, const int* in_sizes, int n_in,
                              void* d_out, int out_size, void* d_ws, size_t ws_size,
                              hipStream_t stream) {
  (void)in_sizes; (void)n_in; (void)out_size; (void)ws_size;
  const float* x  = (const float*)d_in[0];
  const float* wq = (const float*)d_in[1];
  const float* wk = (const float*)d_in[2];
  const float* wv = (const float*)d_in[3];
  const float* wo = (const float*)d_in[4];
  float* out = (float*)d_out;

  char* w = (char*)d_ws;
  size_t off = 0;
  auto carve = [&](size_t bytes) -> char* {
    char* p = w + off;
    off += (bytes + 255) & ~(size_t)255;
    return p;
  };
  bf16*  xb  = (bf16*) carve((size_t)S_LEN * HDIM * 2);
  bf16*  wqb = (bf16*) carve((size_t)HDIM * HDIM * 2);
  bf16*  wkb = (bf16*) carve((size_t)KVD  * HDIM * 2);
  bf16*  wvb = (bf16*) carve((size_t)KVD  * HDIM * 2);
  bf16*  wob = (bf16*) carve((size_t)HDIM * HDIM * 2);
  float* qf  = (float*)carve((size_t)S_LEN * HDIM * 4);
  float* kf  = (float*)carve((size_t)S_LEN * KVD  * 4);
  float* vf  = (float*)carve((size_t)S_LEN * KVD  * 4);
  bf16*  qr  = (bf16*) carve((size_t)S_LEN * HDIM * 2);
  bf16*  kr  = (bf16*) carve((size_t)S_LEN * KVD  * 2);
  bf16*  vt  = (bf16*) carve((size_t)S_LEN * KVD  * 2);
  bf16*  ob  = (bf16*) carve((size_t)S_LEN * HDIM * 2);

  auto cdiv = [](long a, long b) -> unsigned { return (unsigned)((a + b - 1) / b); };

  // 1) casts to bf16
  cast_f32_bf16<<<cdiv((long)S_LEN * HDIM, 256), 256, 0, stream>>>(x,  xb,  (long)S_LEN * HDIM);
  cast_f32_bf16<<<cdiv((long)HDIM * HDIM, 256), 256, 0, stream>>>(wq, wqb, (long)HDIM * HDIM);
  cast_f32_bf16<<<cdiv((long)KVD  * HDIM, 256), 256, 0, stream>>>(wk, wkb, (long)KVD  * HDIM);
  cast_f32_bf16<<<cdiv((long)KVD  * HDIM, 256), 256, 0, stream>>>(wv, wvb, (long)KVD  * HDIM);
  cast_f32_bf16<<<cdiv((long)HDIM * HDIM, 256), 256, 0, stream>>>(wo, wob, (long)HDIM * HDIM);

  // 2) projection GEMMs: grid = (M/128) * (N/64) workgroups
  gemm_bf16<<<16 * 64, 256, 0, stream>>>(xb, wqb, qf, S_LEN, HDIM, HDIM);
  gemm_bf16<<<16 * 16, 256, 0, stream>>>(xb, wkb, kf, S_LEN, KVD,  HDIM);
  gemm_bf16<<<16 * 16, 256, 0, stream>>>(xb, wvb, vf, S_LEN, KVD,  HDIM);

  // 3) RoPE + pack, V transpose
  rope_pack<<<cdiv((long)S_LEN * NHEADS * HEADD, 256), 256, 0, stream>>>(qf, qr, NHEADS);
  rope_pack<<<cdiv((long)S_LEN * NKVH   * HEADD, 256), 256, 0, stream>>>(kf, kr, NKVH);
  v_transpose<<<cdiv((long)NKVH * HEADD * S_LEN, 256), 256, 0, stream>>>(vf, vt);

  // 4) flash attention: NH * (S/16) waves, 8 waves/block
  fa_kernel<<<(NHEADS * (S_LEN / 16)) / 8, 256, 0, stream>>>(qr, kr, vt, ob);

  // 5) output projection into d_out (f32)
  gemm_bf16<<<16 * 64, 256, 0, stream>>>(ob, wob, out, S_LEN, HDIM, HDIM);
}
